// BiRNNDecoder_6485400617612
// MI455X (gfx1250) — matrix operand order repaired and checked by reading, not compiled, not run on hardware
//
#include <hip/hip_runtime.h>
#include <hip/hip_bf16.h>
#include <cstdint>

// ---------------------------------------------------------------------------
// BiRNN decoder with attention for MI455X (gfx1250, wave32, WMMA + TDM).
// Final GEMM uses a double-buffered Tensor-Data-Mover pipeline: A and B tiles
// stream into ping-pong LDS buffers while WMMAs consume the previous pair.
// ---------------------------------------------------------------------------

typedef _Float16 half_t;
typedef __attribute__((ext_vector_type(16))) _Float16 v16h;
typedef __attribute__((ext_vector_type(8)))  float    v8f;
typedef __attribute__((ext_vector_type(4)))  unsigned int u32x4;
typedef __attribute__((ext_vector_type(8)))  int      i32x8;
typedef __attribute__((ext_vector_type(4)))  int      i32x4;

#define WMMA_F32_F16(a, b, c) \
  __builtin_amdgcn_wmma_f32_16x16x32_f16(false, (a), false, (b), (short)0, (c), false, false)

// Tensor Data Mover availability (device pass on gfx1250 toolchains only;
// host pass and other targets fall back to manual LDS staging).
#if defined(__HIP_DEVICE_COMPILE__) && \
    __has_builtin(__builtin_amdgcn_tensor_load_to_lds) && \
    __has_builtin(__builtin_amdgcn_s_wait_tensorcnt)
#define USE_TDM 1
#else
#define USE_TDM 0
#endif

// Problem dims (fixed by the reference)
#define BB   32
#define TT   64
#define SS   64
#define VV   32000
#define EE   512
#define HH   512
#define DD2  1024     // 2*Hs
#define NG   2048     // 4H (gates)
#define KG   1536     // E + H (xin) + H (h) packed K for combined LSTM weight

// ---------------------------------------------------------------------------
// WMMA fragment loaders (contiguous k = 16*(lane/16) + i, same for A and B, wave32)
// ---------------------------------------------------------------------------
__device__ __forceinline__ v16h load_a_frag(const half_t* A, int lda, int m0, int k0) {
  const int lane = threadIdx.x & 31;
  const int row  = m0 + (lane & 15);
  const int kh   = k0 + ((lane >> 4) ? 16 : 0);
  const half_t* p = A + (size_t)row * lda + kh;
  v16h a;
#pragma unroll
  for (int i = 0; i < 16; ++i) a[i] = p[i];
  return a;
}

__device__ __forceinline__ v16h load_b_frag(const half_t* W, int ldw, int n0, int k0) {
  const int lane = threadIdx.x & 31;
  const half_t* p = W + (size_t)(n0 + (lane & 15)) * ldw + k0 + ((lane >> 4) ? 16 : 0);
  v16h b;
#pragma unroll
  for (int i = 0; i < 16; ++i) b[i] = p[i];
  return b;
}

#if USE_TDM
// Issue one TDM 2-D tile load: tile_k x tile_rows (elements) from a row-major
// f16 tensor (row stride = stride0 elements) into LDS at byte offset ldsOff.
// Descriptor layout per CDNA5 ISA ch.8 (D# groups 0/1; groups 2/3 zero = 2D).
__device__ __forceinline__ void tdm_load_2d(const half_t* gptr, unsigned ldsOff,
                                            unsigned tile_k, unsigned tile_rows,
                                            unsigned tensor_dim0, unsigned tensor_dim1,
                                            unsigned stride0) {
  const unsigned long long ga = (unsigned long long)(const void*)gptr;
  u32x4 g0;
  g0[0] = 1u;                                                 // count=1, user D#
  g0[1] = ldsOff;                                             // lds_addr
  g0[2] = (unsigned)(ga & 0xFFFFFFFFu);                       // global_addr[31:0]
  g0[3] = (unsigned)((ga >> 32) & 0x01FFFFFFu) | (2u << 30);  // [56:32] | type=2
  i32x8 g1;
  g1[0] = (int)(1u << 16);                                    // data_size=2B
  g1[1] = (int)((tensor_dim0 & 0xFFFFu) << 16);               // dim0 lo16
  g1[2] = (int)((tensor_dim0 >> 16) | ((tensor_dim1 & 0xFFFFu) << 16));
  g1[3] = (int)((tensor_dim1 >> 16) | (tile_k << 16));        // dim1 hi | tile_dim0
  g1[4] = (int)tile_rows;                                     // tile_dim1 (tile_dim2=0)
  g1[5] = (int)stride0;                                       // dim0 stride lo32
  g1[6] = 0;
  g1[7] = 0;
  i32x4 gz = {0, 0, 0, 0};
#if __clang_major__ >= 23
  i32x8 g4 = {0, 0, 0, 0, 0, 0, 0, 0};
  __builtin_amdgcn_tensor_load_to_lds(g0, g1, gz, gz, g4, 0);
#else
  __builtin_amdgcn_tensor_load_to_lds(g0, g1, gz, gz, 0);
#endif
}
#endif

// ---------------------------------------------------------------------------
// Small recurrent-step GEMM: one wave32 per block, 32(M) x 16(N) output tile.
// A = up to 3 concatenated row-major f16 segments along K; W = f16 [N][K].
// ---------------------------------------------------------------------------
__global__ __launch_bounds__(32) void gemm32(
    const half_t* __restrict__ A0, int lda0,
    const half_t* __restrict__ A1, int lda1,
    const half_t* __restrict__ A2, int lda2,
    int kb0, int kb1, int K,
    const half_t* __restrict__ W,
    const float* __restrict__ bias0, const float* __restrict__ bias1,
    float* __restrict__ OutF, half_t* __restrict__ OutH,
    half_t* __restrict__ OutH2, long ldh2,
    int ldo, int act) {
  const int n0 = blockIdx.x * 16;
  v8f acc0 = {};
  v8f acc1 = {};

  for (int k0 = 0; k0 < kb0; k0 += 32) {
    v16h a0 = load_a_frag(A0, lda0, 0,  k0);
    v16h a1 = load_a_frag(A0, lda0, 16, k0);
    v16h bf = load_b_frag(W, K, n0, k0);
    acc0 = WMMA_F32_F16(a0, bf, acc0);
    acc1 = WMMA_F32_F16(a1, bf, acc1);
  }
  for (int k0 = kb0; k0 < kb1; k0 += 32) {
    v16h a0 = load_a_frag(A1, lda1, 0,  k0 - kb0);
    v16h a1 = load_a_frag(A1, lda1, 16, k0 - kb0);
    v16h bf = load_b_frag(W, K, n0, k0);
    acc0 = WMMA_F32_F16(a0, bf, acc0);
    acc1 = WMMA_F32_F16(a1, bf, acc1);
  }
  for (int k0 = kb1; k0 < K; k0 += 32) {
    v16h a0 = load_a_frag(A2, lda2, 0,  k0 - kb1);
    v16h a1 = load_a_frag(A2, lda2, 16, k0 - kb1);
    v16h bf = load_b_frag(W, K, n0, k0);
    acc0 = WMMA_F32_F16(a0, bf, acc0);
    acc1 = WMMA_F32_F16(a1, bf, acc1);
  }

  const int lane = threadIdx.x & 31;
  const int col  = n0 + (lane & 15);
  const int hi8  = (lane >> 4) * 8;
  float bv = 0.f;
  if (bias0) bv += bias0[col];
  if (bias1) bv += bias1[col];
#pragma unroll
  for (int r = 0; r < 8; ++r) {
    float v0 = acc0[r] + bv;
    float v1 = acc1[r] + bv;
    if (act) { v0 = tanhf(v0); v1 = tanhf(v1); }
    const int m0r = hi8 + r;
    const int m1r = 16 + hi8 + r;
    if (OutF)  { OutF[(size_t)m0r * ldo + col] = v0;
                 OutF[(size_t)m1r * ldo + col] = v1; }
    if (OutH)  { OutH[(size_t)m0r * ldo + col] = (half_t)v0;
                 OutH[(size_t)m1r * ldo + col] = (half_t)v1; }
    if (OutH2) { OutH2[(size_t)m0r * ldh2 + col] = (half_t)v0;
                 OutH2[(size_t)m1r * ldh2 + col] = (half_t)v1; }
  }
}

// ---------------------------------------------------------------------------
// LSTM pointwise: gates g [B,4H] (i,f,g,o chunks of H) + c -> c', h'(f16).
// ---------------------------------------------------------------------------
__global__ void lstm_pw(const float* __restrict__ g, float* __restrict__ c,
                        half_t* __restrict__ h16, const float* __restrict__ mask,
                        int t) {
  int idx = blockIdx.x * 256 + threadIdx.x;
  if (idx >= BB * HH) return;
  const int b = idx >> 9;
  const int n = idx & (HH - 1);
  const float* gb = g + (size_t)b * NG;
  const float gi = gb[n];
  const float gf = gb[HH + n];
  const float gg = gb[2 * HH + n];
  const float go = gb[3 * HH + n];
  const float si = 1.f / (1.f + __expf(-gi));
  const float sf = 1.f / (1.f + __expf(-gf));
  const float so = 1.f / (1.f + __expf(-go));
  float cn = sf * c[idx] + si * tanhf(gg);
  float hn = so * tanhf(cn);
  if (mask) {
    const float m = mask[b * TT + t];
    cn *= m; hn *= m;
  }
  c[idx]  = cn;
  h16[idx] = (half_t)hn;
}

// ---------------------------------------------------------------------------
// Attention: scores = (k . src[b,s]) / sqrt(H), softmax over S,
// ct[b,:] = sum_s p[s] * src[b,s,:].  One block per batch row, S threads.
// ---------------------------------------------------------------------------
__global__ void attn(const float* __restrict__ kbuf, const float* __restrict__ src,
                     half_t* __restrict__ ct16) {
  __shared__ float sw[SS];
  __shared__ float sp[SS];
  const int b = blockIdx.x;
  const int s = threadIdx.x;
  const float* kb = kbuf + (size_t)b * DD2;
  const float* sb = src + ((size_t)b * SS + s) * DD2;
  float acc = 0.f;
  for (int d = 0; d < DD2; ++d) acc += kb[d] * sb[d];
  sw[s] = acc * 0.04419417382415922f;   // 1/sqrt(512)
  __syncthreads();
  float m = -1e30f;
  for (int i = 0; i < SS; ++i) m = fmaxf(m, sw[i]);
  float sum = 0.f;
  for (int i = 0; i < SS; ++i) sum += __expf(sw[i] - m);
  sp[s] = __expf(sw[s] - m) / sum;
  __syncthreads();
#pragma unroll 4
  for (int j = 0; j < DD2 / SS; ++j) {
    const int d = s * (DD2 / SS) + j;
    float a2 = 0.f;
    for (int i = 0; i < SS; ++i) a2 += sp[i] * src[((size_t)b * SS + i) * DD2 + d];
    ct16[(size_t)b * DD2 + d] = (half_t)a2;
  }
}

// ---------------------------------------------------------------------------
// Final projection GEMM: Out[2048,32000] = Xcat[2048,1024](f16) * fcW^T + fc_b.
// 64(M) x 128(N) block tile, BK=32, 256 threads = 8 waves; each wave computes
// a 32x32 subtile (4 WMMA / K-slice).
// TDM path: ping-pong LDS buffers; wave 0 keeps the next A/B tile pair in
// flight on the Tensor Data Mover (in-order per wave), waiting tensorcnt<=2
// so the current pair is resident while the next streams in.
// ---------------------------------------------------------------------------
__global__ __launch_bounds__(256) void gemm_big(
    const half_t* __restrict__ Xc, const half_t* __restrict__ Wv,
    const float* __restrict__ bias, float* __restrict__ Out) {
  __shared__ half_t As[2][64 * 32];
  __shared__ half_t Bs[2][128 * 32];
  const int tid  = threadIdx.x;
  const int n0   = blockIdx.x * 128;
  const int m0   = blockIdx.y * 64;
  const int wave = tid >> 5;
  const int wm   = (wave & 1) * 32;
  const int wn   = (wave >> 1) * 32;
  v8f acc00 = {}, acc01 = {}, acc10 = {}, acc11 = {};

#if USE_TDM
  // Flat LDS addresses carry the LDS byte offset in addr[31:0].
  const unsigned ldsA[2] = {(unsigned)(uintptr_t)(&As[0][0]),
                            (unsigned)(uintptr_t)(&As[1][0])};
  const unsigned ldsB[2] = {(unsigned)(uintptr_t)(&Bs[0][0]),
                            (unsigned)(uintptr_t)(&Bs[1][0])};
  if (tid < 32) {
    tdm_load_2d(Xc + (size_t)m0 * DD2, ldsA[0], 32, 64,  DD2, BB * TT, DD2);
    tdm_load_2d(Wv + (size_t)n0 * DD2, ldsB[0], 32, 128, DD2, VV,      DD2);
  }
  for (int k0 = 0; k0 < DD2; k0 += 32) {
    const int buf = (k0 >> 5) & 1;
    if (tid < 32) {
      if (k0 + 32 < DD2) {
        // Keep the next tile pair in flight while computing the current one.
        tdm_load_2d(Xc + (size_t)m0 * DD2 + k0 + 32, ldsA[buf ^ 1], 32, 64,
                    DD2, BB * TT, DD2);
        tdm_load_2d(Wv + (size_t)n0 * DD2 + k0 + 32, ldsB[buf ^ 1], 32, 128,
                    DD2, VV, DD2);
        __builtin_amdgcn_s_wait_tensorcnt((short)2);  // current pair complete
      } else {
        __builtin_amdgcn_s_wait_tensorcnt((short)0);  // drain on last step
      }
    }
    __syncthreads();

    v16h a0 = load_a_frag(As[buf], 32, wm,      0);
    v16h a1 = load_a_frag(As[buf], 32, wm + 16, 0);
    v16h b0 = load_b_frag(Bs[buf], 32, wn,      0);
    v16h b1 = load_b_frag(Bs[buf], 32, wn + 16, 0);
    acc00 = WMMA_F32_F16(a0, b0, acc00);
    acc01 = WMMA_F32_F16(a0, b1, acc01);
    acc10 = WMMA_F32_F16(a1, b0, acc10);
    acc11 = WMMA_F32_F16(a1, b1, acc11);
    __syncthreads();   // all waves done reading before TDM overwrites this buf
  }
#else
  const int arow = tid >> 2, acg = (tid & 3) * 8;   // A: 64 rows x 32 cols
  const int brow = tid >> 1, bcg = (tid & 1) * 16;  // B: 128 rows x 32 cols
  for (int k0 = 0; k0 < DD2; k0 += 32) {
    {
      const uint4* gp = (const uint4*)(Wv + (size_t)(n0 + brow) * DD2 + k0 + bcg);
      uint4* sp = (uint4*)(Bs[0] + brow * 32 + bcg);
      sp[0] = gp[0];
      sp[1] = gp[1];
    }
    {
      const uint4* gp = (const uint4*)(Xc + (size_t)(m0 + arow) * DD2 + k0 + acg);
      *(uint4*)(As[0] + arow * 32 + acg) = gp[0];
    }
    if (k0 + 32 < DD2) {
      __builtin_prefetch(Xc + (size_t)(m0 + arow) * DD2 + k0 + 32 + acg, 0, 3);
      __builtin_prefetch(Wv + (size_t)(n0 + brow) * DD2 + k0 + 32 + bcg, 0, 3);
    }
    __syncthreads();
    v16h a0 = load_a_frag(As[0], 32, wm,      0);
    v16h a1 = load_a_frag(As[0], 32, wm + 16, 0);
    v16h b0 = load_b_frag(Bs[0], 32, wn,      0);
    v16h b1 = load_b_frag(Bs[0], 32, wn + 16, 0);
    acc00 = WMMA_F32_F16(a0, b0, acc00);
    acc01 = WMMA_F32_F16(a0, b1, acc01);
    acc10 = WMMA_F32_F16(a1, b0, acc10);
    acc11 = WMMA_F32_F16(a1, b1, acc11);
    __syncthreads();
  }
#endif

  const int lane = tid & 31;
  const int colL = lane & 15;
  const int hi8  = (lane >> 4) * 8;
  v8f accs[2][2] = {{acc00, acc01}, {acc10, acc11}};
#pragma unroll
  for (int mi = 0; mi < 2; ++mi) {
#pragma unroll
    for (int ni = 0; ni < 2; ++ni) {
      const int gm = m0 + wm + mi * 16 + hi8;
      const int gn = n0 + wn + ni * 16 + colL;
      const float bv = bias[gn];
#pragma unroll
      for (int r = 0; r < 8; ++r)
        Out[(size_t)(gm + r) * VV + gn] = accs[mi][ni][r] + bv;
    }
  }
}

// ---------------------------------------------------------------------------
// Prep kernels
// ---------------------------------------------------------------------------
__global__ void f2h(const float* __restrict__ in, half_t* __restrict__ out, long n) {
  long i = (long)blockIdx.x * 256 + threadIdx.x;
  if (i < n) out[i] = (half_t)in[i];
}

__global__ void zero_h(half_t* __restrict__ p, long n) {
  long i = (long)blockIdx.x * 256 + threadIdx.x;
  if (i < n) p[i] = (half_t)0.f;
}

// Pack [Wih | Whh] -> f16 [2048][1536]
__global__ void pack_lstm_w(const float* __restrict__ Wih, const float* __restrict__ Whh,
                            half_t* __restrict__ out) {
  long idx = (long)blockIdx.x * 256 + threadIdx.x;
  if (idx >= (long)NG * KG) return;
  const int n = (int)(idx / KG);
  const int k = (int)(idx % KG);
  const float v = (k < EE + HH) ? Wih[(size_t)n * (EE + HH) + k]
                                : Whh[(size_t)n * HH + (k - EE - HH)];
  out[idx] = (half_t)v;
}

// Embedding gather + f16 convert
__global__ void embed_g(const float* __restrict__ emb, const int* __restrict__ trg,
                        half_t* __restrict__ x16) {
  long idx = (long)blockIdx.x * 256 + threadIdx.x;
  if (idx >= (long)BB * TT * EE) return;
  const int e   = (int)(idx % EE);
  const int row = (int)(idx / EE);
  const int tok = trg[row];
  x16[idx] = (half_t)emb[(size_t)tok * EE + e];
}

// Initialize h(f16), c(f32), hhat(f16) from hid_init / feed_init slices
__global__ void init_state(const float* __restrict__ hid, const float* __restrict__ feed,
                           int off_h, int off_c, int off_f,
                           half_t* __restrict__ h16, float* __restrict__ c,
                           half_t* __restrict__ hhat16) {
  int idx = blockIdx.x * 256 + threadIdx.x;
  if (idx >= BB * HH) return;
  const int n = idx & (HH - 1);
  h16[idx]    = (half_t)hid[off_h + n];
  c[idx]      = hid[off_c + n];
  hhat16[idx] = (half_t)feed[off_f + n];
}

// ---------------------------------------------------------------------------
// Host driver
// ---------------------------------------------------------------------------
extern "C" void kernel_launch(void* const* d_in, const int* in_sizes, int n_in,
                              void* d_out, int out_size, void* d_ws, size_t ws_size,
                              hipStream_t stream) {
  (void)in_sizes; (void)n_in; (void)out_size; (void)ws_size;
  const float* src   = (const float*)d_in[0];
  const int*   trg   = (const int*)  d_in[1];
  const float* mask  = (const float*)d_in[2];
  const float* emb   = (const float*)d_in[3];
  const float* fWih  = (const float*)d_in[4];
  const float* fWhh  = (const float*)d_in[5];
  const float* fbih  = (const float*)d_in[6];
  const float* fbhh  = (const float*)d_in[7];
  const float* bWih  = (const float*)d_in[8];
  const float* bWhh  = (const float*)d_in[9];
  const float* bbih  = (const float*)d_in[10];
  const float* bbhh  = (const float*)d_in[11];
  const float* fattW = (const float*)d_in[12];
  const float* fattb = (const float*)d_in[13];
  const float* battW = (const float*)d_in[14];
  const float* battb = (const float*)d_in[15];
  const float* fahW  = (const float*)d_in[16];
  const float* fahb  = (const float*)d_in[17];
  const float* bahW  = (const float*)d_in[18];
  const float* bahb  = (const float*)d_in[19];
  const float* fcW   = (const float*)d_in[20];
  const float* fcb   = (const float*)d_in[21];
  const float* feed  = (const float*)d_in[22];
  const float* hid   = (const float*)d_in[23];
  float* out = (float*)d_out;

  uint8_t* ws = (uint8_t*)d_ws;
  size_t off = 0;
  auto alloc = [&](size_t bytes) -> void* {
    off = (off + 255) & ~(size_t)255;
    void* p = ws + off;
    off += bytes;
    return p;
  };

  half_t* x16    = (half_t*)alloc((size_t)BB * TT * EE * 2);
  half_t* WcF    = (half_t*)alloc((size_t)NG * KG * 2);
  half_t* WcB    = (half_t*)alloc((size_t)NG * KG * 2);
  half_t* fatt16 = (half_t*)alloc((size_t)DD2 * HH * 2);
  half_t* batt16 = (half_t*)alloc((size_t)DD2 * HH * 2);
  half_t* fah16  = (half_t*)alloc((size_t)HH * (HH + DD2) * 2);
  half_t* bah16  = (half_t*)alloc((size_t)HH * (HH + DD2) * 2);
  half_t* fc16   = (half_t*)alloc((size_t)VV * DD2 * 2);
  half_t* Xcat   = (half_t*)alloc((size_t)BB * TT * (2 * HH) * 2);
  half_t* h16    = (half_t*)alloc((size_t)BB * HH * 2);
  half_t* hhat16 = (half_t*)alloc((size_t)BB * HH * 2);
  half_t* ct16   = (half_t*)alloc((size_t)BB * DD2 * 2);
  float*  cbuf   = (float*) alloc((size_t)BB * HH * 4);
  float*  gbuf   = (float*) alloc((size_t)BB * NG * 4);
  float*  kbuf   = (float*) alloc((size_t)BB * DD2 * 4);

  // One-time prep (deterministic every call)
  pack_lstm_w<<<(NG * KG + 255) / 256, 256, 0, stream>>>(fWih, fWhh, WcF);
  pack_lstm_w<<<(NG * KG + 255) / 256, 256, 0, stream>>>(bWih, bWhh, WcB);
  f2h<<<(DD2 * HH + 255) / 256, 256, 0, stream>>>(fattW, fatt16, (long)DD2 * HH);
  f2h<<<(DD2 * HH + 255) / 256, 256, 0, stream>>>(battW, batt16, (long)DD2 * HH);
  f2h<<<(HH * (HH + DD2) + 255) / 256, 256, 0, stream>>>(fahW, fah16, (long)HH * (HH + DD2));
  f2h<<<(HH * (HH + DD2) + 255) / 256, 256, 0, stream>>>(bahW, bah16, (long)HH * (HH + DD2));
  f2h<<<(int)(((long)VV * DD2 + 255) / 256), 256, 0, stream>>>(fcW, fc16, (long)VV * DD2);
  zero_h<<<((long)BB * TT * 2 * HH + 255) / 256, 256, 0, stream>>>(Xcat, (long)BB * TT * 2 * HH);
  embed_g<<<((long)BB * TT * EE + 255) / 256, 256, 0, stream>>>(emb, trg, x16);

  // ---------------- forward direction ----------------
  init_state<<<(BB * HH + 255) / 256, 256, 0, stream>>>(hid, feed, 0, HH, 0,
                                                        h16, cbuf, hhat16);
  for (int t = 0; t < TT; ++t) {
    gemm32<<<NG / 16, 32, 0, stream>>>(
        x16 + (size_t)t * EE, TT * EE, hhat16, HH, h16, HH,
        EE, EE + HH, KG, WcF, fbih, fbhh,
        gbuf, nullptr, nullptr, 0, NG, 0);
    lstm_pw<<<(BB * HH + 255) / 256, 256, 0, stream>>>(gbuf, cbuf, h16, nullptr, 0);
    gemm32<<<DD2 / 16, 32, 0, stream>>>(
        h16, HH, h16, HH, h16, HH,
        HH, HH, HH, fatt16, fattb, nullptr,
        kbuf, nullptr, nullptr, 0, DD2, 0);
    attn<<<BB, SS, 0, stream>>>(kbuf, src, ct16);
    gemm32<<<HH / 16, 32, 0, stream>>>(
        h16, HH, ct16, DD2, ct16, DD2,
        HH, HH + DD2, HH + DD2, bah16, bahb, nullptr,
        nullptr, hhat16, Xcat + (size_t)t * (2 * HH), (long)TT * 2 * HH, HH, 1);
  }

  // ---------------- backward direction ----------------
  init_state<<<(BB * HH + 255) / 256, 256, 0, stream>>>(hid, feed, 2 * HH, 3 * HH, HH,
                                                        h16, cbuf, hhat16);
  for (int tb = 0; tb < TT; ++tb) {
    const int ti = TT - 1 - tb;
    gemm32<<<NG / 16, 32, 0, stream>>>(
        x16 + (size_t)ti * EE, TT * EE, hhat16, HH, h16, HH,
        EE, EE + HH, KG, WcB, bbih, bbhh,
        gbuf, nullptr, nullptr, 0, NG, 0);
    lstm_pw<<<(BB * HH + 255) / 256, 256, 0, stream>>>(gbuf, cbuf, h16, mask, ti);
    gemm32<<<DD2 / 16, 32, 0, stream>>>(
        h16, HH, h16, HH, h16, HH,
        HH, HH, HH, batt16, battb, nullptr,
        kbuf, nullptr, nullptr, 0, DD2, 0);
    attn<<<BB, SS, 0, stream>>>(kbuf, src, ct16);
    half_t* x2 = (tb >= 2) ? (Xcat + (size_t)(tb - 2) * (2 * HH) + HH) : nullptr;
    gemm32<<<HH / 16, 32, 0, stream>>>(
        h16, HH, ct16, DD2, ct16, DD2,
        HH, HH + DD2, HH + DD2, fah16, fahb, nullptr,
        nullptr, hhat16, x2, (long)TT * 2 * HH, HH, 1);
  }

  // ---------------- output projection ----------------
  gemm_big<<<dim3(VV / 128, (BB * TT) / 64), 256, 0, stream>>>(Xcat, fc16, fcb, out);
}
